// PrescriptionPill_54872502174378
// MI455X (gfx1250) — compile-verified, matrix operand order checked by reading, not run on hardware
//
#include <hip/hip_runtime.h>
#include <math.h>

typedef __attribute__((ext_vector_type(16))) __bf16 v16bf;
typedef __attribute__((ext_vector_type(8)))  float  v8f;

#define D_DIM 2048
#define P_DIM 256
#define NSEG  64
#define LCAP  1024

__device__ __forceinline__ int lower_bound_i32(const int* __restrict__ a, int n, int v) {
    int lo = 0, hi = n;
    while (lo < hi) { int mid = (lo + hi) >> 1; if (a[mid] < v) lo = mid + 1; else hi = mid; }
    return lo;
}

// ---------------------------------------------------------------------------
// K0: convert W1/W2 (f32 [256,2048]) into bf16 hi/lo WMMA B-fragments.
// Fragment layout (16-bit B 32x16, K x N), per lane L: N = L%16, element t
// holds K = kc*32 + (t/8)*16 + (L/16)*8 + (t%8). Stored so each lane reads
// 16 contiguous bf16 (32B) per fragment. frag id = m*1024 + ct*64 + kc.
// ---------------------------------------------------------------------------
__global__ void prep_w_kernel(const float* __restrict__ W1, const float* __restrict__ W2,
                              __bf16* __restrict__ fhi, __bf16* __restrict__ flo) {
    int frag = blockIdx.x;          // 0..2047
    int lane = threadIdx.x;         // 0..31
    int m  = frag >> 10;
    int ct = (frag >> 6) & 15;
    int kc = frag & 63;
    const float* W = m ? W2 : W1;
    int p    = ct * 16 + (lane & 15);
    int half = lane >> 4;
    size_t base = ((size_t)frag * 32 + lane) * 16;
#pragma unroll
    for (int t = 0; t < 16; ++t) {
        int k = kc * 32 + (t >> 3) * 16 + half * 8 + (t & 7);
        float w = W[(size_t)p * D_DIM + k];
        __bf16 h = (__bf16)w;
        __bf16 l = (__bf16)(w - (float)h);
        fhi[base + t] = h;
        flo[base + t] = l;
    }
}

// ---------------------------------------------------------------------------
// K1: dual GEMM  Y1 = x@W1^T, Y2 = x@W2^T  (no bias), bf16 hi/lo split.
// Block: 256 threads = 8 waves, 64-row x 256-col tile.
// Wave w: matrix m=w&1, row strip=(w>>1)*16. 16 column tiles, v8f acc each.
// A fragment (16-bit A 16x32): lane L: M=L%16, element t holds
// K = kc*32 + (t/8)*16 + (L/16)*8 + (t%8)  -> two 8-float runs per lane.
// ---------------------------------------------------------------------------
__global__ void __launch_bounds__(256, 1)
gemm_dual_kernel(const float* __restrict__ x,
                 const __bf16* __restrict__ fhi, const __bf16* __restrict__ flo,
                 float* __restrict__ Y1, float* __restrict__ Y2) {
    const int wv    = threadIdx.x >> 5;
    const int lane  = threadIdx.x & 31;
    const int m     = wv & 1;
    const int strip = wv >> 1;
    const int half  = lane >> 4;
    const int rbase = blockIdx.x * 64 + strip * 16;
    const int row   = rbase + (lane & 15);
    const float4* xr = reinterpret_cast<const float4*>(x + (size_t)row * D_DIM);

    v8f acc[16] = {};

    for (int kc = 0; kc < 64; ++kc) {
        int i0 = kc * 8 + half * 2;
        float4 a0 = xr[i0],     a1 = xr[i0 + 1];
        float4 a2 = xr[i0 + 4], a3 = xr[i0 + 5];
        float f[16] = { a0.x, a0.y, a0.z, a0.w,  a1.x, a1.y, a1.z, a1.w,
                        a2.x, a2.y, a2.z, a2.w,  a3.x, a3.y, a3.z, a3.w };
        v16bf ah = {}, al = {};
#pragma unroll
        for (int t = 0; t < 16; ++t) {
            __bf16 h = (__bf16)f[t];
            ah[t] = h;
            al[t] = (__bf16)(f[t] - (float)h);
        }
#pragma unroll
        for (int ct = 0; ct < 16; ++ct) {
            size_t fb = (((size_t)(m * 1024 + ct * 64 + kc)) * 32 + lane) * 16;
            v16bf bh = *reinterpret_cast<const v16bf*>(fhi + fb);
            v16bf bl = *reinterpret_cast<const v16bf*>(flo + fb);
            acc[ct] = __builtin_amdgcn_wmma_f32_16x16x32_bf16(false, ah, false, bh,
                                                              (short)0, acc[ct], false, false);
            acc[ct] = __builtin_amdgcn_wmma_f32_16x16x32_bf16(false, al, false, bh,
                                                              (short)0, acc[ct], false, false);
            acc[ct] = __builtin_amdgcn_wmma_f32_16x16x32_bf16(false, ah, false, bl,
                                                              (short)0, acc[ct], false, false);
        }
    }

    float* Y = m ? Y2 : Y1;
#pragma unroll
    for (int ct = 0; ct < 16; ++ct) {
#pragma unroll
        for (int v = 0; v < 8; ++v) {
            int r = rbase + v + 8 * half;             // C layout: VGPR v -> M = v + 8*(lane/16)
            Y[(size_t)r * P_DIM + ct * 16 + (lane & 15)] = acc[ct][v];
        }
    }
}

// ---------------------------------------------------------------------------
// K2: per-segment mean of x  ->  seg_mean [64, 2048]
// ---------------------------------------------------------------------------
__global__ void segmean_kernel(const float* __restrict__ x, const int* __restrict__ lbatch,
                               float* __restrict__ segmean, int N) {
    __shared__ int sS, sE;
    if (threadIdx.x == 0) {
        sS = lower_bound_i32(lbatch, N, (int)blockIdx.x);
        sE = lower_bound_i32(lbatch, N, (int)blockIdx.x + 1);
    }
    __syncthreads();
    int s = sS, len = sE - sS;
    float inv = (len > 0) ? 1.0f / (float)len : 0.0f;
    for (int d = threadIdx.x; d < D_DIM; d += 256) {
        float a = 0.0f;
        for (int i = 0; i < len; ++i) a += x[(size_t)(s + i) * D_DIM + d];
        segmean[blockIdx.x * D_DIM + d] = a * inv;
    }
}

// ---------------------------------------------------------------------------
// K3: projected = seg_mean @ proj_w^T + proj_b   [64, 256]
// ---------------------------------------------------------------------------
__global__ void proj_kernel(const float* __restrict__ segmean, const float* __restrict__ pw,
                            const float* __restrict__ pb, float* __restrict__ projected) {
    __shared__ float sm[D_DIM];
    int b = blockIdx.x, p = threadIdx.x;
    for (int d = p; d < D_DIM; d += 256) sm[d] = segmean[b * D_DIM + d];
    __syncthreads();
    float acc = pb[p];
    const float4* wr = reinterpret_cast<const float4*>(pw + (size_t)p * D_DIM);
    for (int d4 = 0; d4 < D_DIM / 4; ++d4) {
        float4 w = wr[d4];
        acc += sm[d4 * 4 + 0] * w.x + sm[d4 * 4 + 1] * w.y +
               sm[d4 * 4 + 2] * w.z + sm[d4 * 4 + 3] * w.w;
    }
    projected[b * P_DIM + p] = acc;
}

// ---------------------------------------------------------------------------
// K4: h = gelu(projected) @ fc_w^T + fc_b + projected; LayerNorm -> projOut
// ---------------------------------------------------------------------------
__global__ void head_kernel(const float* __restrict__ projected, const float* __restrict__ fw,
                            const float* __restrict__ fb, const float* __restrict__ lng,
                            const float* __restrict__ lnb, float* __restrict__ projOut) {
    __shared__ float g[P_DIM];
    __shared__ float red[P_DIM];
    int b = blockIdx.x, p = threadIdx.x;
    float pr = projected[b * P_DIM + p];
    float t  = 0.7978845608028654f * (pr + 0.044715f * pr * pr * pr);
    g[p] = 0.5f * pr * (1.0f + tanhf(t));        // jax.nn.gelu (approximate=True)
    __syncthreads();
    float acc = fb[p];
    const float4* wr = reinterpret_cast<const float4*>(fw + (size_t)p * P_DIM);
    for (int q4 = 0; q4 < P_DIM / 4; ++q4) {
        float4 w = wr[q4];
        acc += g[q4 * 4 + 0] * w.x + g[q4 * 4 + 1] * w.y +
               g[q4 * 4 + 2] * w.z + g[q4 * 4 + 3] * w.w;
    }
    float h = acc + pr;
    red[p] = h; __syncthreads();
    for (int off = 128; off > 0; off >>= 1) { if (p < off) red[p] += red[p + off]; __syncthreads(); }
    float mu = red[0] * (1.0f / P_DIM);
    __syncthreads();
    red[p] = (h - mu) * (h - mu); __syncthreads();
    for (int off = 128; off > 0; off >>= 1) { if (p < off) red[p] += red[p + off]; __syncthreads(); }
    float var = red[0] * (1.0f / P_DIM);
    projOut[b * P_DIM + p] = (h - mu) * rsqrtf(var + 1e-5f) * lng[p] + lnb[p];
}

// ---------------------------------------------------------------------------
// K5: per-segment leave-label-out assembly + broadcast of projOut.
// Rank-sort rows by label in LDS so same-label runs are contiguous; then
// out1[i,p] = Y1[i,p] + W1_b[p] + W2_b[p] + (tot[p]-runSum[p]) / other_cnt
// out2[i,p] = projOut[b,p]
// ---------------------------------------------------------------------------
__global__ void loo_kernel(const float* __restrict__ Y1, const float* __restrict__ Y2,
                           const float* __restrict__ projOut,
                           const int* __restrict__ label, const int* __restrict__ lbatch,
                           const float* __restrict__ b1, const float* __restrict__ b2,
                           float* __restrict__ out1, float* __restrict__ out2, int N) {
    __shared__ int sS, sE;
    __shared__ int   labs[LCAP];
    __shared__ short order[LCAP];
    int b = blockIdx.x, p = threadIdx.x;
    if (p == 0) {
        sS = lower_bound_i32(lbatch, N, b);
        sE = lower_bound_i32(lbatch, N, b + 1);
    }
    __syncthreads();
    int start = sS;
    int len = sE - start;
    if (len > LCAP) len = LCAP;
    for (int j = p; j < len; j += 256) labs[j] = label[start + j];
    __syncthreads();
    // parallel rank sort by (label, index)
    for (int j = p; j < len; j += 256) {
        int lj = labs[j]; int r = 0;
        for (int k2 = 0; k2 < len; ++k2) {
            int lk = labs[k2];
            r += (lk < lj) || (lk == lj && k2 < j);
        }
        order[r] = (short)j;
    }
    __syncthreads();
    float totp = 0.0f;
    for (int j = 0; j < len; ++j) totp += Y2[(size_t)(start + j) * P_DIM + p];
    float bias = b1[p] + b2[p];
    float po   = projOut[b * P_DIM + p];
    int rs = 0;
    while (rs < len) {                              // uniform control flow (LDS-driven)
        int l0 = labs[order[rs]];
        int re = rs + 1;
        while (re < len && labs[order[re]] == l0) ++re;
        float same = 0.0f;
        for (int r = rs; r < re; ++r) same += Y2[(size_t)(start + order[r]) * P_DIM + p];
        int othc = len - (re - rs);
        float looval = (othc > 0) ? (totp - same) / (float)othc : 0.0f;
        for (int r = rs; r < re; ++r) {
            size_t o = (size_t)(start + order[r]) * P_DIM + p;
            out1[o] = Y1[o] + bias + looval;
            out2[o] = po;
        }
        rs = re;
    }
}

extern "C" void kernel_launch(void* const* d_in, const int* in_sizes, int n_in,
                              void* d_out, int out_size, void* d_ws, size_t ws_size,
                              hipStream_t stream) {
    const float* x      = (const float*)d_in[0];
    const int*   label  = (const int*)  d_in[1];
    const int*   lbatch = (const int*)  d_in[2];
    const float* W1     = (const float*)d_in[3];
    const float* W1b    = (const float*)d_in[4];
    const float* W2     = (const float*)d_in[5];
    const float* W2b    = (const float*)d_in[6];
    const float* pw     = (const float*)d_in[7];
    const float* pb     = (const float*)d_in[8];
    const float* fw     = (const float*)d_in[9];
    const float* fb     = (const float*)d_in[10];
    const float* lng    = (const float*)d_in[11];
    const float* lnb    = (const float*)d_in[12];

    const int N = in_sizes[0] / D_DIM;             // 8192

    char* ws = (char*)d_ws;
    __bf16* fhi      = reinterpret_cast<__bf16*>(ws);                       // 2 MB
    __bf16* flo      = reinterpret_cast<__bf16*>(ws + (size_t)(2  << 20));  // 2 MB
    float*  Y1       = reinterpret_cast<float*> (ws + (size_t)(4  << 20));  // 8 MB
    float*  Y2       = reinterpret_cast<float*> (ws + (size_t)(12 << 20));  // 8 MB
    float*  segmean  = reinterpret_cast<float*> (ws + (size_t)(20 << 20));  // 512 KB
    float*  projected = segmean + (size_t)NSEG * D_DIM;                     // 64 KB
    float*  projOut   = projected + (size_t)NSEG * P_DIM;                   // 64 KB

    float* out1 = (float*)d_out;
    float* out2 = out1 + (size_t)N * P_DIM;

    prep_w_kernel   <<<2048,  32, 0, stream>>>(W1, W2, fhi, flo);
    gemm_dual_kernel<<<N / 64, 256, 0, stream>>>(x, fhi, flo, Y1, Y2);
    segmean_kernel  <<<NSEG, 256, 0, stream>>>(x, lbatch, segmean, N);
    proj_kernel     <<<NSEG, 256, 0, stream>>>(segmean, pw, pb, projected);
    head_kernel     <<<NSEG, 256, 0, stream>>>(projected, fw, fb, lng, lnb, projOut);
    loo_kernel      <<<NSEG, 256, 0, stream>>>(Y1, Y2, projOut, label, lbatch,
                                               W1b, W2b, out1, out2, N);
}
